// RDeformConv2d_25606595019208
// MI455X (gfx1250) — matrix operand order repaired and checked by reading, not compiled, not run on hardware
//
#include <hip/hip_runtime.h>
#include <hip/hip_bf16.h>

typedef __attribute__((ext_vector_type(16))) __bf16 v16bf;
typedef __attribute__((ext_vector_type(8)))  float  v8f;

#define BATCH   8
#define INCH    128
#define OUTCH   128
#define HH      64
#define WW      64
#define HW      4096          // 64*64
#define KTOT    1152          // 128*9
#define KSTEPS  36            // 1152/32
#define NT1     2             // 27 offset+mask channels -> 2 n-tiles (padded to 32)
#define NT2     8             // 128 out channels -> 8 n-tiles
#define B1_ELEMS (KSTEPS * NT1 * 32 * 16)   // 36864
#define B2_ELEMS (KSTEPS * NT2 * 32 * 16)   // 147456

// A-matrix 16x32 bf16 fragment: lane holds M=lane%16; element e covers
// K = (e<8 ? e : e+8) + 8*(lane/16)   (ISA 7.12.2, 16-bit A layout)
__device__ __forceinline__ int ka_lo(int e) { return (e < 8) ? e : e + 8; }
// B-matrix 32x16 bf16 fragment: lane holds N=lane%16; element e covers
// K = (e<8 ? e : e+8) + 8*(lane/16)
__device__ __forceinline__ int kb_idx(int e, int lane) {
  return ((e < 8) ? e : e + 8) + ((lane >> 4) << 3);
}

// gfx1250 async global->LDS fill (VGLOBAL opcode 98, tracked by ASYNCcnt)
__device__ __forceinline__ void async_copy_b128(unsigned lds_byte_addr, const void* gptr) {
  asm volatile("global_load_async_to_lds_b128 %0, %1, off"
               :: "v"(lds_byte_addr), "v"(gptr) : "memory");
}
__device__ __forceinline__ void wait_async0() {
  asm volatile("s_wait_asynccnt 0" ::: "memory");
}

// -------------------------------------------------------------------------
// Kernel 0: pack all weights into WMMA B-fragment order, f32 -> bf16.
//   b1: [36 ksteps][2 ntiles][32 lanes][16 elems]  (p_w ++ m_w, padded to 32ch)
//   b2: [36 ksteps][8 ntiles][32 lanes][16 elems]  (conv_w)
// K ordering: K = t*128 + c  (kernel-tap major)
// -------------------------------------------------------------------------
__global__ void pack_weights(const float* __restrict__ p_w,
                             const float* __restrict__ m_w,
                             const float* __restrict__ conv_w,
                             __bf16* __restrict__ b1,
                             __bf16* __restrict__ b2) {
  int id = blockIdx.x * blockDim.x + threadIdx.x;
  if (id < B1_ELEMS) {
    int e = id & 15, lane = (id >> 4) & 31, nt = (id >> 9) & 1, ks = id >> 10;
    int K = ks * 32 + kb_idx(e, lane);
    int t = K / 128, c = K & 127;
    int n = nt * 16 + (lane & 15);
    float v = 0.f;
    if (n < 18)      v = p_w[(size_t)n * KTOT + c * 9 + t];
    else if (n < 27) v = m_w[(size_t)(n - 18) * KTOT + c * 9 + t];
    b1[id] = (__bf16)v;
  } else if (id < B1_ELEMS + B2_ELEMS) {
    int id2 = id - B1_ELEMS;
    int e = id2 & 15, lane = (id2 >> 4) & 31, nt = (id2 >> 9) & 7, ks = id2 >> 12;
    int K = ks * 32 + kb_idx(e, lane);
    int t = K / 128, c = K & 127;
    int n = nt * 16 + (lane & 15);
    b2[id2] = (__bf16)conv_w[(size_t)n * KTOT + c * 9 + t];
  }
}

// -------------------------------------------------------------------------
// Kernel 1: offset(18ch)+mask(9ch) 3x3 conv as implicit WMMA GEMM.
// One wave = one 16-pixel M-tile (consecutive j in one row), K loop = 36
// (fully unrolled: tap/channel indices constant-fold, loads get immediate
// offsets off one base pointer). Out-of-image taps: clamp index, zero weight.
// obuf layout: [b][ch 0..26][4096 pixels], mask channels pre-sigmoided.
// -------------------------------------------------------------------------
__global__ void __launch_bounds__(128)
offset_mask_gemm(const float* __restrict__ x,
                 const __bf16* __restrict__ b1,
                 const float* __restrict__ p_b,
                 const float* __restrict__ m_b,
                 float* __restrict__ obuf) {
  const int lane = threadIdx.x & 31;
  const int wave = threadIdx.x >> 5;
  const int tile = blockIdx.x * 4 + wave;        // 2048 tiles
  const int pixbase = tile << 4;
  const int b   = pixbase >> 12;
  const int pix0 = pixbase & 4095;
  const int i   = pix0 >> 6;
  const int jm  = (pix0 & 63) + (lane & 15);     // this lane's pixel column
  // base pointer including the lane-dependent K-half offset: loads below use
  // only compile-time channel immediates on top of this.
  const float* xh = x + (size_t)b * INCH * HW + (size_t)((lane >> 4) << 3) * HW;

  const v8f vzero = {0.f,0.f,0.f,0.f,0.f,0.f,0.f,0.f};
  v8f acc0 = vzero, acc1 = vzero;

  const v16bf* bfr = (const v16bf*)b1;
#pragma unroll
  for (int ks = 0; ks < KSTEPS; ++ks) {
    const int t  = ks >> 2;              // constant after unroll
    const int c0 = (ks & 3) << 5;
    const int ii = i + (t / 3) - 1;
    const int jj = jm + (t % 3) - 1;
    const bool valid = (ii >= 0) & (ii < HH) & (jj >= 0) & (jj < WW);
    const float wmul = valid ? 1.f : 0.f;
    const int spc = min(max(ii, 0), HH - 1) * WW + min(max(jj, 0), WW - 1);

    v16bf a;
#pragma unroll
    for (int e = 0; e < 16; ++e) {
      const int clo = c0 + ka_lo(e);     // compile-time constant
      a[e] = (__bf16)(wmul * xh[(size_t)clo * HW + spc]);
    }
    v16bf bf0 = bfr[(size_t)(ks * NT1 + 0) * 32 + lane];
    v16bf bf1 = bfr[(size_t)(ks * NT1 + 1) * 32 + lane];
    acc0 = __builtin_amdgcn_wmma_f32_16x16x32_bf16(false, a, false, bf0, (short)0, acc0, false, false);
    acc1 = __builtin_amdgcn_wmma_f32_16x16x32_bf16(false, a, false, bf1, (short)0, acc1, false, false);
  }

  // C layout: lane -> N = lane%16; VGPR r -> M = r + 8*(lane/16)
  const int nlo = lane & 15;
  const int mhi = (lane >> 4) << 3;
#pragma unroll
  for (int nt = 0; nt < 2; ++nt) {
    const v8f acc = nt ? acc1 : acc0;
    const int ch = nt * 16 + nlo;
    if (ch >= 27) continue;
    const float bias = (ch < 18) ? p_b[ch] : m_b[ch - 18];
#pragma unroll
    for (int r = 0; r < 8; ++r) {
      float v = acc[r] + bias;
      if (ch >= 18) v = 1.f / (1.f + __expf(-v));   // sigmoid for mask channels
      obuf[(size_t)(b * 27 + ch) * HW + pix0 + mhi + r] = v;
    }
  }
}

// -------------------------------------------------------------------------
// Kernel 2: fused bilinear sampling + 128-ch output GEMM.
// out[b,o,i,j] = sum_{c,t} conv_w[o,c,t] * x_off[b,c,i,j,t]
// One wave = 16-pixel M-tile, all 8 N-tiles (8 v8f accumulators).
// Per tap t: bilinear coeffs/indices once per lane, reused for 4 K-steps;
// out-of-image corners get clamped indices + zeroed weights (no exec
// divergence). B fragments staged block-wide via async global->LDS b128.
// -------------------------------------------------------------------------
__global__ void __launch_bounds__(128)
deform_gemm(const float* __restrict__ x,
            const float* __restrict__ obuf,
            const __bf16* __restrict__ b2,
            float* __restrict__ out) {
  __shared__ __attribute__((aligned(32))) __bf16 lds_b[NT2 * 32 * 16];  // 8192 B

  const int lane = threadIdx.x & 31;
  const int wave = threadIdx.x >> 5;
  const int tile = blockIdx.x * 4 + wave;        // 2048 tiles
  const int pixbase = tile << 4;
  const int b    = pixbase >> 12;
  const int pix0 = pixbase & 4095;
  const int i    = pix0 >> 6;
  const int jm   = (pix0 & 63) + (lane & 15);
  const int pixm = pix0 + (lane & 15);

  const float* xh = x + (size_t)b * INCH * HW + (size_t)((lane >> 4) << 3) * HW;
  const float* ob = obuf + (size_t)b * 27 * HW;
  const unsigned lds_base = (unsigned)(size_t)(&lds_b[0]);

  const v8f vzero = {0.f,0.f,0.f,0.f,0.f,0.f,0.f,0.f};
  v8f acc[NT2];
#pragma unroll
  for (int nt = 0; nt < NT2; ++nt) acc[nt] = vzero;

  for (int t = 0; t < 9; ++t) {
    // --- per-lane bilinear setup on the 66x66 zero-padded grid ---
    const float mval = ob[(size_t)(18 + t) * HW + pixm];
    const float px = ob[(size_t)t * HW + pixm]       + (float)(i + 1)  + (float)(t / 3 - 1);
    const float py = ob[(size_t)(9 + t) * HW + pixm] + (float)(jm + 1) + (float)(t % 3 - 1);
    const float fx = floorf(px), fy = floorf(py);
    const int ltx = min(max((int)fx, 0), 65);
    const int lty = min(max((int)fy, 0), 65);
    const int rbx = min(max((int)fx + 1, 0), 65);
    const int rby = min(max((int)fy + 1, 0), 65);
    const float pxc = fminf(fmaxf(px, 0.f), 65.f);
    const float pyc = fminf(fmaxf(py, 0.f), 65.f);
    const float gx_l = 1.f + (float)ltx - pxc;
    const float gx_r = 1.f - ((float)rbx - pxc);
    const float gy_l = 1.f + (float)lty - pyc;
    const float gy_r = 1.f - ((float)rby - pyc);
    // padded coord q maps to x[q-1]; out-of-image taps: weight->0, index clamped
    const bool vx_l = (ltx >= 1) & (ltx <= 64), vx_r = (rbx >= 1) & (rbx <= 64);
    const bool vy_l = (lty >= 1) & (lty <= 64), vy_r = (rby >= 1) & (rby <= 64);
    const float g_lt = (vx_l & vy_l) ? gx_l * gy_l * mval : 0.f;
    const float g_rb = (vx_r & vy_r) ? gx_r * gy_r * mval : 0.f;
    const float g_lb = (vx_l & vy_r) ? gx_l * gy_r * mval : 0.f;
    const float g_rt = (vx_r & vy_l) ? gx_r * gy_l * mval : 0.f;
    const int o_lt = min(max((ltx - 1) * WW + (lty - 1), 0), HW - 1);
    const int o_rb = min(max((rbx - 1) * WW + (rby - 1), 0), HW - 1);
    const int o_lb = min(max((ltx - 1) * WW + (rby - 1), 0), HW - 1);
    const int o_rt = min(max((rbx - 1) * WW + (lty - 1), 0), HW - 1);

#pragma unroll
    for (int cb = 0; cb < 4; ++cb) {
      const int ks = t * 4 + cb;

      // stage this kstep's 8 B fragments (8 KB) into LDS via async b128 fills
      __syncthreads();
      {
        const __bf16* src = b2 + (size_t)ks * NT2 * 32 * 16;
#pragma unroll
        for (int q = 0; q < 4; ++q) {
          const unsigned chunk = threadIdx.x + q * 128;     // 16B chunks
          async_copy_b128(lds_base + chunk * 16, src + chunk * 8);
        }
        if (ks + 1 < KSTEPS)
          __builtin_prefetch(b2 + (size_t)(ks + 1) * NT2 * 32 * 16, 0, 1);
        wait_async0();
      }
      __syncthreads();

      // build A fragment: 32 channels x 16 pixels of sampled x_off
      const int c0 = cb << 5;
      v16bf a;
      if (t == 4) {                       // center tap: raw x, no mask
#pragma unroll
        for (int e = 0; e < 16; ++e) {
          const int clo = c0 + ka_lo(e);
          a[e] = (__bf16)xh[(size_t)clo * HW + pixm];
        }
      } else {
#pragma unroll
        for (int e = 0; e < 16; ++e) {
          const int clo = c0 + ka_lo(e);
          const float* xc = xh + (size_t)clo * HW;
          float s = g_lt * xc[o_lt];
          s += g_rb * xc[o_rb];
          s += g_lb * xc[o_lb];
          s += g_rt * xc[o_rt];
          a[e] = (__bf16)s;
        }
      }

      const v16bf* bl = (const v16bf*)lds_b;
#pragma unroll
      for (int nt = 0; nt < NT2; ++nt) {
        v16bf bf = bl[nt * 32 + lane];
        acc[nt] = __builtin_amdgcn_wmma_f32_16x16x32_bf16(false, a, false, bf, (short)0, acc[nt], false, false);
      }
    }
  }

  // store C tiles: out is (B, 128, 64, 64) f32
  const int nlo = lane & 15;
  const int mhi = (lane >> 4) << 3;
#pragma unroll
  for (int nt = 0; nt < NT2; ++nt) {
    const int ch = nt * 16 + nlo;
#pragma unroll
    for (int r = 0; r < 8; ++r)
      out[(size_t)(b * OUTCH + ch) * HW + pix0 + mhi + r] = acc[nt][r];
  }
}

// -------------------------------------------------------------------------
extern "C" void kernel_launch(void* const* d_in, const int* in_sizes, int n_in,
                              void* d_out, int out_size, void* d_ws, size_t ws_size,
                              hipStream_t stream) {
  const float* x      = (const float*)d_in[0];
  const float* p_w    = (const float*)d_in[1];
  const float* p_b    = (const float*)d_in[2];
  const float* m_w    = (const float*)d_in[3];
  const float* m_b    = (const float*)d_in[4];
  const float* conv_w = (const float*)d_in[5];
  float* out = (float*)d_out;

  // workspace: packed weights (bf16) + offset/mask conv output (f32) ~3.9 MB
  __bf16* b1   = (__bf16*)d_ws;                 // 36864 elems
  __bf16* b2   = b1 + B1_ELEMS;                 // 147456 elems
  float*  obuf = (float*)(b2 + B2_ELEMS);       // 8*27*4096 floats

  // 184320 pack elements
  pack_weights<<<720, 256, 0, stream>>>(p_w, m_w, conv_w, b1, b2);
  // 2048 M-tiles, 4 waves/block
  offset_mask_gemm<<<512, 128, 0, stream>>>(x, b1, p_b, m_b, obuf);
  deform_gemm<<<512, 128, 0, stream>>>(x, obuf, b2, out);
}
